// BackboneCSAI_38732015075464
// MI455X (gfx1250) — matrix operand order repaired and checked by reading, not compiled, not run on hardware
//
#include <hip/hip_runtime.h>
#include <math.h>

#define B_ 256
#define T_ 64
#define F_ 128
#define H_ 512
#define C_ 256
#define HEADS_ 8
#define DH_ 32
#define T2_ 128
#define DFF_ 64
#define M1_ (B_*T_)
#define M2_ (B_*T2_)
#define C3_ (3*C_)

typedef __attribute__((ext_vector_type(16))) __bf16 v16bf;
typedef __attribute__((ext_vector_type(8)))  __bf16 v8bf;
typedef __attribute__((ext_vector_type(8)))  float  v8f;

__device__ __forceinline__ v8f wmma_bf16(v16bf a, v16bf b, v8f c) {
  // D = A(16x32) * B(32x16) + C, f32 accumulate
  return __builtin_amdgcn_wmma_f32_16x16x32_bf16(false, a, false, b, (short)0, c, false, false);
}

// B-fragment: two 8-wide bf16 runs along K at p and p+16 (lane col = lane&15, K offset hi*8)
__device__ __forceinline__ v16bf load_b_bf16(const __bf16* p) {
  v8bf b0 = *(const v8bf*)(p);
  v8bf b1 = *(const v8bf*)(p + 16);
  v16bf b;
#pragma unroll
  for (int j = 0; j < 8; ++j) { b[j] = b0[j]; b[8 + j] = b1[j]; }
  return b;
}

// C = A[M,K](f32) @ W[N,K](bf16)^T + bias.
// Block = 4 waves sharing one 16-row M strip; the A tile (16x64) is cooperatively
// converted to bf16 and staged in LDS ONCE per block (4x less A traffic + cvt),
// then each wave reads its fragment as two ds_load_b128 and computes a 16x64 tile.
// EPI: 0=bias  1=bias+exact GELU  2=bias+PE add w/ [B,T,C]->[B,2T,C] row remap
//      3=exp(-relu(v+bias))      4=bias, bf16 output
template<int EPI>
__global__ __launch_bounds__(128)
void gemm_wmma_kernel(const float* __restrict__ A, int lda,
                      const __bf16* __restrict__ W,
                      const float* __restrict__ bias,
                      float* __restrict__ Cout, int ldc,
                      int M, int N, int K,
                      const float* __restrict__ pe, int tOff) {
  __shared__ __bf16 As[16 * 64];       // one 16(M) x 64(K) bf16 tile, 2KB
  const int tid  = threadIdx.x;
  const int lane = tid & 31;
  const int wave = tid >> 5;
  const int col  = lane & 15;
  const int hi   = lane >> 4;
  const int nBase0 = (blockIdx.x * 4 + wave) * 64;
  const int mBase = blockIdx.y * 16;
  int rem = nBase0 < N ? (N - nBase0) >> 4 : 0;
  const int nt = rem < 4 ? rem : 4;    // waves with nt==0 still help stage + barrier
  v8f acc[4];
#pragma unroll
  for (int i = 0; i < 4; ++i) acc[i] = (v8f){0.f,0.f,0.f,0.f,0.f,0.f,0.f,0.f};
  // staging coords: thread stages 8 consecutive k of one row
  const int srow = tid >> 3;           // 0..15
  const int skof = (tid & 7) * 8;      // 0,8,..,56
  const float*  sap = A + (size_t)(mBase + srow) * lda + skof;
  const __bf16* wr0 = W + (size_t)((nt > 0 ? nBase0 : 0) + col) * K + hi * 8;
  for (int kb = 0; kb < K; kb += 64) {
    const float4 f0 = *(const float4*)(sap + kb);
    const float4 f1 = *(const float4*)(sap + kb + 4);
    __builtin_prefetch(sap + kb + 64, 0, 1);
    v8bf s;
    s[0]=(__bf16)f0.x; s[1]=(__bf16)f0.y; s[2]=(__bf16)f0.z; s[3]=(__bf16)f0.w;
    s[4]=(__bf16)f1.x; s[5]=(__bf16)f1.y; s[6]=(__bf16)f1.z; s[7]=(__bf16)f1.w;
    __syncthreads();                   // previous chunk fully consumed
    *(v8bf*)(&As[srow * 64 + skof]) = s;
    __syncthreads();                   // chunk visible to all waves
#pragma unroll
    for (int ks = 0; ks < 2; ++ks) {
      const int k = kb + ks * 32;
      const __bf16* ap = &As[col * 64 + ks * 32 + hi * 8];
      v8bf a0 = *(const v8bf*)(ap);
      v8bf a1 = *(const v8bf*)(ap + 16);
      v16bf af;
#pragma unroll
      for (int j = 0; j < 8; ++j) { af[j] = a0[j]; af[8 + j] = a1[j]; }
#pragma unroll
      for (int i = 0; i < 4; ++i) {
        if (i < nt) {
          v16bf bw = load_b_bf16(wr0 + (size_t)(i * 16) * K + k);
          acc[i] = wmma_bf16(af, bw, acc[i]);
        }
      }
    }
  }
#pragma unroll
  for (int i = 0; i < 4; ++i) {
    if (i >= nt) break;
#pragma unroll
    for (int r = 0; r < 8; ++r) {
      int m = mBase + r + hi * 8;
      int n = nBase0 + i * 16 + col;
      float v = acc[i][r] + bias[n];
      if (EPI == 1) v = 0.5f * v * (1.f + erff(v * 0.70710678118654752f));
      if (EPI == 3) v = expf(-fmaxf(v, 0.f));
      if (EPI == 2) {
        v += pe[(m % T_) * C_ + n];
        int outRow = (m / T_) * T2_ + (m % T_) + tOff;
        Cout[(size_t)outRow * ldc + n] = v;
      } else if (EPI == 4) {
        ((__bf16*)Cout)[(size_t)m * ldc + n] = (__bf16)v;
      } else {
        Cout[(size_t)m * ldc + n] = v;
      }
    }
  }
}

// Fused batch-axis attention on bf16 qkv: per (n,h) and 16-row s-tile, one wave.
// scores(16x256) via 16 WMMA (K=DH=32, zero-cvt bf16 loads), scaled at LDS store,
// softmax over t in LDS, then attn@V with the LDS A-fragment shared by both d-tiles.
__global__ __launch_bounds__(32)
void attn_kernel(const __bf16* __restrict__ qkv, float* __restrict__ ao) {
  const int lane = threadIdx.x;
  const int nh = blockIdx.x;           // 0..T2_*HEADS_-1
  const int n = nh >> 3, hH = nh & 7;
  const int sTile = blockIdx.y;        // 0..15
  const int col = lane & 15, hi = lane >> 4;
  __shared__ float sm[16 * 256];
  const float scale = 0.1767766952966369f;  // 1/sqrt(DH)
  v16bf aq;
  {
    const int srow = sTile * 16 + col;
    const __bf16* qp = qkv + ((size_t)srow * T2_ + n) * C3_ + hH * DH_ + hi * 8;
    v8bf q0 = *(const v8bf*)(qp);
    v8bf q1 = *(const v8bf*)(qp + 16);
#pragma unroll
    for (int j = 0; j < 8; ++j) { aq[j] = q0[j]; aq[8 + j] = q1[j]; }
  }
  for (int tt = 0; tt < 16; ++tt) {
    const int tcol = tt * 16 + col;
    v16bf bk = load_b_bf16(qkv + ((size_t)tcol * T2_ + n) * C3_ + C_ + hH * DH_ + hi * 8);
    v8f acc = {0.f,0.f,0.f,0.f,0.f,0.f,0.f,0.f};
    acc = wmma_bf16(aq, bk, acc);
#pragma unroll
    for (int r = 0; r < 8; ++r) sm[(r + hi * 8) * 256 + tt * 16 + col] = acc[r] * scale;
  }
  __syncthreads();
  // softmax: lane pair (L, L+16) share row L&15; each scans 128 cols
  const int row = col;
  float mx = -3.0e38f;
  for (int c = 0; c < 128; ++c) mx = fmaxf(mx, sm[row * 256 + hi * 128 + c]);
  mx = fmaxf(mx, __shfl_xor(mx, 16, 32));
  float sum = 0.f;
  for (int c = 0; c < 128; ++c) {
    int idx = row * 256 + hi * 128 + c;
    float e = expf(sm[idx] - mx);
    sm[idx] = e;
    sum += e;
  }
  sum += __shfl_xor(sum, 16, 32);
  float rinv = 1.f / sum;
  __syncthreads();
  v8f acc0 = {0.f,0.f,0.f,0.f,0.f,0.f,0.f,0.f};
  v8f acc1 = {0.f,0.f,0.f,0.f,0.f,0.f,0.f,0.f};
  for (int k = 0; k < 256; k += 32) {
    const float* ap = sm + (size_t)row * 256 + k + hi * 8;
    v16bf aa;
#pragma unroll
    for (int j = 0; j < 16; ++j) aa[j] = (__bf16)(ap[((j >> 3) << 4) + (j & 7)] * rinv);
    v16bf bv0, bv1;
#pragma unroll
    for (int j = 0; j < 16; ++j) {
      size_t base = ((size_t)(k + hi * 8 + ((j >> 3) << 4) + (j & 7)) * T2_ + n) * C3_ + 2 * C_ + hH * DH_ + col;
      bv0[j] = qkv[base];
      bv1[j] = qkv[base + 16];
    }
    acc0 = wmma_bf16(aa, bv0, acc0);
    acc1 = wmma_bf16(aa, bv1, acc1);
  }
#pragma unroll
  for (int r = 0; r < 8; ++r) {
    int s = sTile * 16 + r + hi * 8;
    ao[((size_t)s * T2_ + n) * C_ + hH * DH_ + col] = acc0[r];
    ao[((size_t)s * T2_ + n) * C_ + hH * DH_ + 16 + col] = acc1[r];
  }
}

__global__ void cvt_bf16_kernel(const float* __restrict__ src, __bf16* __restrict__ dst, int n) {
  int i = blockIdx.x * blockDim.x + threadIdx.x;
  if (i < n) dst[i] = (__bf16)src[i];
}

__global__ void prep_kernel(float* pe, const float* w_dx, float* dx_diag,
                            const float* w_fr, __bf16* wb_frm, float* loss_acc) {
  int i = blockIdx.x * blockDim.x + threadIdx.x;  // 0..16383
  if (i < T_ * C_) {
    int t = i / C_, c = i % C_;
    int ci = c >> 1;
    float div = expf(-9.210340371976184f * (2.0f * (float)ci) / (float)C_);
    float ang = (float)t * div;
    pe[i] = (c & 1) ? cosf(ang) : sinf(ang);
  }
  if (i < F_) dx_diag[i] = w_dx[i * F_ + i];
  if (i < F_ * F_) {
    int r = i / F_, c = i % F_;
    wb_frm[i] = (r == c) ? (__bf16)0.f : (__bf16)w_fr[i];
  }
  if (i < 2 * T_) loss_acc[i] = 0.f;
}

__global__ void dd_kernel(const float* __restrict__ deltas, const float* __restrict__ medians,
                          float* __restrict__ dd) {
  int i = blockIdx.x * blockDim.x + threadIdx.x;  // < M1_*F_
  dd[i] = deltas[i] - medians[i % F_];
}

__global__ void decay_elt_kernel(const float* __restrict__ deltas, const float* __restrict__ medians,
                                 const float* __restrict__ lin, float* __restrict__ outp) {
  int i = blockIdx.x * blockDim.x + threadIdx.x;  // < M1_*F_
  float dv = deltas[i] - medians[i % F_];
  float s = (dv > 0.f) ? 1.f : ((dv < 0.f) ? -1.f : 0.f);
  outp[i] = 0.5f * (1.f - tanhf(s * fabsf(lin[i])));
}

// out = LayerNorm(a + b) * g + be ; one wave per row of 256
__global__ __launch_bounds__(32)
void add_ln_kernel(const float* __restrict__ a, const float* __restrict__ b,
                   const float* __restrict__ g, const float* __restrict__ be,
                   float* __restrict__ outp) {
  const int row = blockIdx.x, lane = threadIdx.x;
  float v[8]; float s = 0.f;
#pragma unroll
  for (int j = 0; j < 8; ++j) {
    int c = lane + j * 32;
    float t = a[(size_t)row * C_ + c] + b[(size_t)row * C_ + c];
    v[j] = t; s += t;
  }
  for (int off = 16; off > 0; off >>= 1) s += __shfl_xor(s, off, 32);
  float mu = s * (1.f / (float)C_);
  float var = 0.f;
#pragma unroll
  for (int j = 0; j < 8; ++j) { float d = v[j] - mu; var += d * d; }
  for (int off = 16; off > 0; off >>= 1) var += __shfl_xor(var, off, 32);
  var *= (1.f / (float)C_);
  float r = rsqrtf(var + 1e-5f);
#pragma unroll
  for (int j = 0; j < 8; ++j) {
    int c = lane + j * 32;
    outp[(size_t)row * C_ + c] = (v[j] - mu) * r * g[c] + be[c];
  }
}

__global__ void h0_kernel(const float* __restrict__ proj, const float* __restrict__ w_op2,
                          const float* __restrict__ b_op2, float* __restrict__ h) {
  int i = blockIdx.x * blockDim.x + threadIdx.x;  // B_*H_
  int b = i / H_, hh = i % H_;
  float s = b_op2[0];
  for (int t = 0; t < T2_; ++t) s += proj[((size_t)b * T2_ + t) * H_ + hh] * w_op2[t];
  h[i] = s;
}

__global__ void hmul_kernel(float* __restrict__ h, const float* __restrict__ g) {
  int i = blockIdx.x * blockDim.x + threadIdx.x;
  h[i] *= g[i];
}

__global__ void xr_cat1_kernel(const float* __restrict__ x_t, int lda,
                               const float* __restrict__ m_t, const float* __restrict__ d_t,
                               const float* __restrict__ x_h, const float* __restrict__ dx_diag,
                               const float* __restrict__ b_dx,
                               float* __restrict__ x_r, float* __restrict__ cat1) {
  int i = blockIdx.x * blockDim.x + threadIdx.x;  // B_*F_
  int b = i / F_, f = i % F_;
  float xv = x_t[(size_t)b * lda + f];
  float mv = m_t[(size_t)b * lda + f];
  float dv = d_t[(size_t)b * lda + f];
  x_r[i] = mv * xv + (1.f - mv) * x_h[i];
  float gx = expf(-fmaxf(dv * dx_diag[f] + b_dx[f], 0.f));
  cat1[(size_t)b * (2 * F_) + f] = gx;
  cat1[(size_t)b * (2 * F_) + F_ + f] = mv;
}

__global__ void xcomb_kernel(const float* __restrict__ x_t, int lda,
                             const float* __restrict__ m_t,
                             const float* __restrict__ beta, const float* __restrict__ xu,
                             const float* __restrict__ x_h,
                             float* __restrict__ cat2,
                             float* __restrict__ ximp_out, float* __restrict__ rec_out,
                             float* __restrict__ lnum, float* __restrict__ lden) {
  int i = blockIdx.x * blockDim.x + threadIdx.x;  // B_*F_
  int b = i / F_, f = i % F_;
  float bt = beta[i];
  float xc = bt * xu[i] + (1.f - bt) * x_h[i];
  float xv = x_t[(size_t)b * lda + f];
  float mv = m_t[(size_t)b * lda + f];
  float xi = mv * xv + (1.f - mv) * xc;
  ximp_out[(size_t)b * lda + f] = xi;
  rec_out[(size_t)b * lda + f] = xc;
  cat2[(size_t)b * (2 * F_) + f] = xi;
  cat2[(size_t)b * (2 * F_) + F_ + f] = mv;
  float ns = fabsf(xv - xc) * mv, ds = mv;
  for (int off = 16; off > 0; off >>= 1) {
    ns += __shfl_xor(ns, off, 32);
    ds += __shfl_xor(ds, off, 32);
  }
  if ((threadIdx.x & 31) == 0) { atomicAdd(lnum, ns); atomicAdd(lden, ds); }
}

__global__ void gru_kernel(const float* __restrict__ gi, const float* __restrict__ gh,
                           float* __restrict__ h) {
  int i = blockIdx.x * blockDim.x + threadIdx.x;  // B_*H_
  int b = i / H_, hh = i % H_;
  size_t base = (size_t)b * (3 * H_);
  float r = 1.f / (1.f + expf(-(gi[base + hh] + gh[base + hh])));
  float z = 1.f / (1.f + expf(-(gi[base + H_ + hh] + gh[base + H_ + hh])));
  float nn = tanhf(gi[base + 2 * H_ + hh] + r * gh[base + 2 * H_ + hh]);
  h[i] = (1.f - z) * nn + z * h[i];
}

__global__ void final_kernel(const float* __restrict__ loss_acc, const float* __restrict__ h,
                             float* __restrict__ out_h, float* __restrict__ out_loss) {
  int i = blockIdx.x * blockDim.x + threadIdx.x;
  if (i < B_ * H_) out_h[i] = h[i];
  if (i == 0) {
    float s = 0.f;
    for (int t = 0; t < T_; ++t) s += loss_acc[2 * t] / (loss_acc[2 * t + 1] + 1e-5f);
    *out_loss = s;
  }
}

#define LAUNCH_GEMM(EPI, Aptr, LDA, Wptr, BIAS, Cptr, LDC, M, N, K, PE, TOFF)           \
  gemm_wmma_kernel<EPI><<<dim3(((N) + 255) / 256, (M) / 16), dim3(128), 0, stream>>>(   \
      Aptr, LDA, Wptr, BIAS, Cptr, LDC, M, N, K, PE, TOFF)

#define LAUNCH_CVT(SRC, DST, N) \
  cvt_bf16_kernel<<<dim3(((N) + 255) / 256), dim3(256), 0, stream>>>(SRC, DST, N)

extern "C" void kernel_launch(void* const* d_in, const int* in_sizes, int n_in,
                              void* d_out, int out_size, void* d_ws, size_t ws_size,
                              hipStream_t stream) {
  (void)in_sizes; (void)n_in; (void)out_size; (void)ws_size;
  const float* x        = (const float*)d_in[0];
  const float* mask     = (const float*)d_in[1];
  const float* deltas   = (const float*)d_in[2];
  const float* last_obs = (const float*)d_in[3];
  const float* medians  = (const float*)d_in[4];
  const float* w_obs    = (const float*)d_in[5];
  const float* b_obs    = (const float*)d_in[6];
  const float* w_in     = (const float*)d_in[7];
  const float* b_in     = (const float*)d_in[8];
  const float* wqkv     = (const float*)d_in[9];
  const float* bqkv     = (const float*)d_in[10];
  const float* wo       = (const float*)d_in[11];
  const float* bo       = (const float*)d_in[12];
  const float* ln1g     = (const float*)d_in[13];
  const float* ln1b     = (const float*)d_in[14];
  const float* w_ff1    = (const float*)d_in[15];
  const float* b_ff1    = (const float*)d_in[16];
  const float* w_ff2    = (const float*)d_in[17];
  const float* b_ff2    = (const float*)d_in[18];
  const float* ln2g     = (const float*)d_in[19];
  const float* ln2b     = (const float*)d_in[20];
  const float* w_op1    = (const float*)d_in[21];
  const float* b_op1    = (const float*)d_in[22];
  const float* w_op2    = (const float*)d_in[23];
  const float* b_op2    = (const float*)d_in[24];
  const float* w_dh     = (const float*)d_in[25];
  const float* b_dh     = (const float*)d_in[26];
  const float* w_dx     = (const float*)d_in[27];
  const float* b_dx     = (const float*)d_in[28];
  const float* w_hist   = (const float*)d_in[29];
  const float* b_hist   = (const float*)d_in[30];
  const float* w_fr     = (const float*)d_in[31];
  const float* b_fr     = (const float*)d_in[32];
  const float* w_wc     = (const float*)d_in[33];
  const float* b_wc     = (const float*)d_in[34];
  const float* w_ih     = (const float*)d_in[35];
  const float* b_ih     = (const float*)d_in[36];
  const float* w_hh     = (const float*)d_in[37];
  const float* b_hh     = (const float*)d_in[38];

  float* ws  = (float*)d_ws;
  float* out = (float*)d_out;

  // region layout (floats), reused across phases (~177 MB total)
  float* R0 = ws + 0;         // 8,388,608  : dd -> data -> ff2
  float* R1 = ws + 8388608;   // 25,165,824 : dec_tmp+decay -> qkv(bf16) -> woout+src+ff1 -> proj
  float* R2 = ws + 33554432;  // 8,388,608  : ao -> src2
  float* S  = ws + 41943040;  // smalls + bf16 weight arena
  float* ddbuf  = R0;
  float* data   = R0;
  float* ff2    = R0;
  float* dec_tmp= R1;
  float* decay  = R1 + 2097152;
  __bf16* qkv_bf = (__bf16*)R1;          // 25,165,824 bf16 = 12.58M float slots
  float* woout  = R1;
  float* src    = R1 + 8388608;
  float* ff1    = R1 + 16777216;
  float* proj   = R1;
  float* ao     = R2;
  float* src2   = R2;

  float* pe      = S;             // 16384
  float* dx_diag = S + 16384;     // 128
  float* loss_acc= S + 16512;     // 128
  float* h       = S + 16640;     // 131072
  float* gamma_h = S + 147712;    // 131072
  float* x_h     = S + 278784;    // 32768
  float* x_r     = S + 311552;    // 32768
  float* xu      = S + 344320;    // 32768
  float* cat1    = S + 377088;    // 65536
  float* beta    = S + 442624;    // 32768
  float* cat2    = S + 475392;    // 65536
  float* gi      = S + 540928;    // 393216
  float* gh      = S + 934144;    // 393216
  __bf16* wb = (__bf16*)(S + 1327360);   // bf16 weight arena (1,835,008 elems)
  __bf16* wb_obs  = wb;            // 16384
  __bf16* wb_in   = wb + 16384;    // 32768
  __bf16* wb_qkv  = wb + 49152;    // 196608
  __bf16* wb_o    = wb + 245760;   // 65536
  __bf16* wb_ff1  = wb + 311296;   // 16384
  __bf16* wb_ff2  = wb + 327680;   // 16384
  __bf16* wb_op1  = wb + 344064;   // 131072
  __bf16* wb_dh   = wb + 475136;   // 65536
  __bf16* wb_hist = wb + 540672;   // 65536
  __bf16* wb_frm  = wb + 606208;   // 16384 (w_fr with zeroed diagonal)
  __bf16* wb_wc   = wb + 622592;   // 32768
  __bf16* wb_ih   = wb + 655360;   // 393216
  __bf16* wb_hh   = wb + 1048576;  // 786432

  float* out_ximp = out;               // [B,T,F]
  float* out_rec  = out + 2097152;     // [B,T,F]
  float* out_h    = out + 4194304;     // [B,H]
  float* out_loss = out + 4325376;     // scalar

  // ---- prep: PE table, diag(w_dx), masked bf16 w_fr, zero loss; bf16 weight conversion ----
  prep_kernel<<<dim3(64), dim3(256), 0, stream>>>(pe, w_dx, dx_diag, w_fr, wb_frm, loss_acc);
  LAUNCH_CVT(w_obs,  wb_obs,  F_ * F_);
  LAUNCH_CVT(w_in,   wb_in,   C_ * F_);
  LAUNCH_CVT(wqkv,   wb_qkv,  C3_ * C_);
  LAUNCH_CVT(wo,     wb_o,    C_ * C_);
  LAUNCH_CVT(w_ff1,  wb_ff1,  DFF_ * C_);
  LAUNCH_CVT(w_ff2,  wb_ff2,  C_ * DFF_);
  LAUNCH_CVT(w_op1,  wb_op1,  H_ * C_);
  LAUNCH_CVT(w_dh,   wb_dh,   H_ * F_);
  LAUNCH_CVT(w_hist, wb_hist, F_ * H_);
  LAUNCH_CVT(w_wc,   wb_wc,   F_ * 2 * F_);
  LAUNCH_CVT(w_ih,   wb_ih,   3 * H_ * 2 * F_);
  LAUNCH_CVT(w_hh,   wb_hh,   3 * H_ * H_);

  // ---- decay_obs branch ----
  dd_kernel<<<dim3(8192), dim3(256), 0, stream>>>(deltas, medians, ddbuf);
  LAUNCH_GEMM(0, ddbuf, F_, wb_obs, b_obs, dec_tmp, F_, M1_, F_, F_, (const float*)nullptr, 0);
  decay_elt_kernel<<<dim3(8192), dim3(256), 0, stream>>>(deltas, medians, dec_tmp, decay);

  // ---- data = concat(last_obs @ w_in^T + b + pe, decay @ w_in^T + b + pe) ----
  LAUNCH_GEMM(2, last_obs, F_, wb_in, b_in, data, C_, M1_, C_, F_, pe, 0);
  LAUNCH_GEMM(2, decay,    F_, wb_in, b_in, data, C_, M1_, C_, F_, pe, T_);

  // ---- transformer layer (qkv produced directly in bf16 for the attention kernel) ----
  LAUNCH_GEMM(4, data, C_, wb_qkv, bqkv, (float*)qkv_bf, C3_, M2_, C3_, C_, (const float*)nullptr, 0);
  attn_kernel<<<dim3(T2_ * HEADS_, B_ / 16), dim3(32), 0, stream>>>(qkv_bf, ao);
  LAUNCH_GEMM(0, ao, C_, wb_o, bo, woout, C_, M2_, C_, C_, (const float*)nullptr, 0);
  add_ln_kernel<<<dim3(M2_), dim3(32), 0, stream>>>(data, woout, ln1g, ln1b, src);
  LAUNCH_GEMM(1, src, C_, wb_ff1, b_ff1, ff1, DFF_, M2_, DFF_, C_, (const float*)nullptr, 0);
  LAUNCH_GEMM(0, ff1, DFF_, wb_ff2, b_ff2, ff2, C_, M2_, C_, DFF_, (const float*)nullptr, 0);
  add_ln_kernel<<<dim3(M2_), dim3(32), 0, stream>>>(src, ff2, ln2g, ln2b, src2);

  // ---- h0 projection ----
  LAUNCH_GEMM(0, src2, C_, wb_op1, b_op1, proj, H_, M2_, H_, C_, (const float*)nullptr, 0);
  h0_kernel<<<dim3(512), dim3(256), 0, stream>>>(proj, w_op2, b_op2, h);

  // ---- recurrent imputation scan (T_ sequential steps) ----
  for (int t = 0; t < T_; ++t) {
    const float* x_t = x + (size_t)t * F_;
    const float* m_t = mask + (size_t)t * F_;
    const float* d_t = deltas + (size_t)t * F_;
    const int lda = T_ * F_;
    LAUNCH_GEMM(3, d_t, lda, wb_dh, b_dh, gamma_h, H_, B_, H_, F_, (const float*)nullptr, 0);
    hmul_kernel<<<dim3(512), dim3(256), 0, stream>>>(h, gamma_h);
    LAUNCH_GEMM(0, h, H_, wb_hist, b_hist, x_h, F_, B_, F_, H_, (const float*)nullptr, 0);
    xr_cat1_kernel<<<dim3(128), dim3(256), 0, stream>>>(x_t, lda, m_t, d_t, x_h, dx_diag, b_dx, x_r, cat1);
    LAUNCH_GEMM(0, x_r, F_, wb_frm, b_fr, xu, F_, B_, F_, F_, (const float*)nullptr, 0);
    LAUNCH_GEMM(0, cat1, 2 * F_, wb_wc, b_wc, beta, F_, B_, F_, 2 * F_, (const float*)nullptr, 0);
    xcomb_kernel<<<dim3(128), dim3(256), 0, stream>>>(x_t, lda, m_t, beta, xu, x_h, cat2,
                                                      out_ximp + (size_t)t * F_, out_rec + (size_t)t * F_,
                                                      loss_acc + 2 * t, loss_acc + 2 * t + 1);
    LAUNCH_GEMM(0, cat2, 2 * F_, wb_ih, b_ih, gi, 3 * H_, B_, 3 * H_, 2 * F_, (const float*)nullptr, 0);
    LAUNCH_GEMM(0, h, H_, wb_hh, b_hh, gh, 3 * H_, B_, 3 * H_, H_, (const float*)nullptr, 0);
    gru_kernel<<<dim3(512), dim3(256), 0, stream>>>(gi, gh, h);
  }

  final_kernel<<<dim3(512), dim3(256), 0, stream>>>(loss_acc, h, out_h, out_loss);
}
